// FusedLlamaLayerIntra_80908593922054
// MI455X (gfx1250) — matrix operand order repaired and checked by reading, not compiled, not run on hardware
//
#include <hip/hip_runtime.h>
#include <hip/hip_bf16.h>

// ---------------------------------------------------------------------------
// Fused quant-LoRA LLaMA layer for gfx1250 (CDNA5, wave32, WMMA).
// All matmuls go through v_wmma_f32_16x16x32_f16.
// GEMM: double-buffered LDS, software-pipelined global loads, 4 WMMA/K-step.
// ---------------------------------------------------------------------------

typedef __attribute__((ext_vector_type(16))) _Float16 v16h;
typedef __attribute__((ext_vector_type(8)))  float    v8f;

// Problem constants (match reference)
constexpr int kS  = 1024;
constexpr int kC  = 2048;
constexpr int kH  = 32;
constexpr int kD  = 64;
constexpr int kHKV = 8;
constexpr int kKV = kHKV * kD;   // 512
constexpr int kR  = 16;
constexpr int kFF = 5632;
constexpr int kNO = 20;          // max(1, int(0.01*2048))
constexpr float kQMaxInv = 1.0f / 7.0f;   // Q_BIT=4 -> qmax=7

// Output layout (floats, concatenated reference tuple)
constexpr long N_XOUT   = (long)kS * kC;
constexpr long O_XIDX   = N_XOUT;
constexpr long O_XSCALE = O_XIDX + kNO;
constexpr long O_XN1S   = O_XSCALE + kC;
constexpr long O_QS     = O_XN1S + kC;
constexpr long O_KS     = O_QS + kD;
constexpr long O_VS     = O_KS + kD;
constexpr long O_ATHR   = O_VS + kD;
constexpr long O_OS     = O_ATHR + 1;
constexpr long O_XMIDX  = O_OS + kC;
constexpr long O_XMSCALE= O_XMIDX + kNO;
constexpr long O_XN2S   = O_XMSCALE + kC;
constexpr long O_GS     = O_XN2S + kC;
constexpr long O_US     = O_GS + kFF;
constexpr long O_FNS    = O_US + kFF;
constexpr long O_HADS   = O_FNS + kFF;

// ---------------------------------------------------------------------------
// WMMA fragment helpers (16x16x32 f16, contiguous k order shared by A and B, not the ISA table order)
// ---------------------------------------------------------------------------
static __device__ __forceinline__ v8f wmma32(v16h a, v16h b, v8f c) {
  return __builtin_amdgcn_wmma_f32_16x16x32_f16(false, a, false, b, (short)0, c,
                                                false, false);
}

// A-matrix 16x32: element A[m][k] at base[m*mstride + k] (k pairs contiguous).
// lanes 0-15: M=lane, K=0..15 (2 per VGPR); lanes 16-31: K=16..31.
static __device__ __forceinline__ v16h frag_a_mk(const _Float16* base, int mstride) {
  const int lane = threadIdx.x & 31;
  const int m = lane & 15, hi = lane >> 4;
  v16h f;
  unsigned int* fp = reinterpret_cast<unsigned int*>(&f);
#pragma unroll
  for (int j = 0; j < 8; ++j) {
    int k = (hi << 4) + (j << 1);
    fp[j] = *reinterpret_cast<const unsigned int*>(base + (size_t)m * mstride + k);
  }
  return f;
}

// B-matrix 32x16: element B[k][n] at base[n*nstride + k] (k contiguous per n).
// lanes 0-15: N=lane, K=0..15 (2 per VGPR); lanes 16-31: K=16..31.
static __device__ __forceinline__ v16h frag_b_nk(const _Float16* base, int nstride) {
  const int lane = threadIdx.x & 31;
  const int n = lane & 15, hi = lane >> 4;
  v16h f;
  unsigned int* fp = reinterpret_cast<unsigned int*>(&f);
#pragma unroll
  for (int j = 0; j < 8; ++j) {
    int k = (hi << 4) + (j << 1);
    fp[j] = *reinterpret_cast<const unsigned int*>(base + (size_t)n * nstride + k);
  }
  return f;
}

// B-matrix 32x16 gathered: element B[k][n] at base[k*kstride + n] (k-major memory).
static __device__ __forceinline__ v16h frag_b_gather(const _Float16* base, int kstride) {
  const int lane = threadIdx.x & 31;
  const int n = lane & 15, hi = lane >> 4;
  v16h f;
#pragma unroll
  for (int j = 0; j < 8; ++j) {
    int k = (hi << 4) + (j << 1);
    f[2 * j]     = base[(size_t)k * kstride + n];
    f[2 * j + 1] = base[(size_t)(k + 1) * kstride + n];
  }
  return f;
}

// ---------------------------------------------------------------------------
// Weight dequant: w_f16[o][i] = wi[o][i] * sc[o]
// ---------------------------------------------------------------------------
__global__ void dequant_w(const int* __restrict__ wi, const float* __restrict__ sc,
                          _Float16* __restrict__ out, size_t total, int in_d) {
  size_t i = (size_t)blockIdx.x * 256 + threadIdx.x;
  size_t stride = (size_t)gridDim.x * 256;
  for (; i < total; i += stride)
    out[i] = (_Float16)((float)wi[i] * sc[i / in_d]);
}

// ---------------------------------------------------------------------------
// RMSNorm: one block per row
// ---------------------------------------------------------------------------
__global__ __launch_bounds__(256)
void rmsnorm_f16(const float* __restrict__ X, const float* __restrict__ w,
                 _Float16* __restrict__ out, int C) {
  __shared__ float red[8];
  __shared__ float rstd_s;
  const int row = blockIdx.x;
  const float* xr = X + (size_t)row * C;
  float ss = 0.0f;
  for (int c = threadIdx.x; c < C; c += 256) {
    float v = xr[c];
    ss += v * v;
  }
#pragma unroll
  for (int off = 16; off; off >>= 1) ss += __shfl_xor(ss, off, 32);
  if ((threadIdx.x & 31) == 0) red[threadIdx.x >> 5] = ss;
  __syncthreads();
  if (threadIdx.x == 0) {
    float t = 0.0f;
    for (int j = 0; j < 8; ++j) t += red[j];
    rstd_s = rsqrtf(t / C + 1e-5f);
  }
  __syncthreads();
  const float rstd = rstd_s;
  for (int c = threadIdx.x; c < C; c += 256)
    out[(size_t)row * C + c] = (_Float16)(xr[c] * rstd * w[c]);
}

// ---------------------------------------------------------------------------
// LoRA down-projection xa[m][r] = sum_k X[m][k] * la[k][r]   (R=16)
// ---------------------------------------------------------------------------
__global__ void lora_xa(const _Float16* __restrict__ X, const float* __restrict__ la,
                        float* __restrict__ xa, int M, int K) {
  int t = blockIdx.x * 256 + threadIdx.x;
  if (t >= M * kR) return;
  int m = t >> 4, r = t & 15;
  float s = 0.0f;
  for (int k = 0; k < K; ++k)
    s += (float)X[(size_t)m * K + k] * la[(size_t)k * kR + r];
  xa[t] = s;
}

// ---------------------------------------------------------------------------
// WMMA GEMM: Y[M,N] = A[M,K] @ W[N,K]^T + bias[N] + xa[M,16] @ lb[16,N]
// block = 256 thr (8 waves), block tile 64(M) x 128(N), wave tile 32x32,
// K stepped by 32 through double-buffered LDS; global loads for tile t+1
// overlap the 4 WMMAs of tile t; one barrier per K-step.
// ---------------------------------------------------------------------------
__global__ __launch_bounds__(256)
void wmma_gemm_f16(const _Float16* __restrict__ A, const _Float16* __restrict__ W,
                   const float* __restrict__ bias, const float* __restrict__ xa,
                   const float* __restrict__ lb, float* __restrict__ Y,
                   int M, int N, int K) {
  __shared__ _Float16 sA[2][64][40];
  __shared__ _Float16 sB[2][128][40];
  const int tid = threadIdx.x;
  const int m0 = blockIdx.y * 64;
  const int n0 = blockIdx.x * 128;
  const int wv = tid >> 5, lane = tid & 31;
  const int wm = (wv >> 2) * 32;          // 0 or 32
  const int wn = (wv & 3) * 32;           // 0,32,64,96
  const int rA = (tid * 8) >> 5, cA = (tid * 8) & 31;    // A: 8 halves/thread
  const int rB = (tid * 16) >> 5, cB = (tid * 16) & 31;  // B: 16 halves/thread
  v8f acc00 = {}, acc01 = {}, acc10 = {}, acc11 = {};
  const int nk = K >> 5;
  uint4 ra, rb0, rb1;
  {  // preload K-tile 0
    const _Float16* ga = A + (size_t)(m0 + rA) * K + cA;
    const _Float16* gb = W + (size_t)(n0 + rB) * K + cB;
    ra  = *(const uint4*)ga;
    rb0 = *(const uint4*)gb;
    rb1 = *(const uint4*)(gb + 8);
  }
  *(uint4*)&sA[0][rA][cA] = ra;
  *(uint4*)&sB[0][rB][cB] = rb0;
  *(uint4*)&sB[0][rB][cB + 8] = rb1;
  __syncthreads();
  for (int kt = 0; kt < nk; ++kt) {
    const int buf = kt & 1;
    if (kt + 1 < nk) {   // issue next tile's global loads before the math
      const int kb = (kt + 1) * 32;
      const _Float16* ga = A + (size_t)(m0 + rA) * K + kb + cA;
      const _Float16* gb = W + (size_t)(n0 + rB) * K + kb + cB;
      ra  = *(const uint4*)ga;
      rb0 = *(const uint4*)gb;
      rb1 = *(const uint4*)(gb + 8);
      __builtin_prefetch(ga + 32, 0, 3);
      __builtin_prefetch(gb + 32, 0, 3);
    }
    v16h a0 = frag_a_mk(&sA[buf][wm][0], 40);
    v16h a1 = frag_a_mk(&sA[buf][wm + 16][0], 40);
    v16h b0 = frag_b_nk(&sB[buf][wn][0], 40);
    v16h b1 = frag_b_nk(&sB[buf][wn + 16][0], 40);
    acc00 = wmma32(a0, b0, acc00);
    acc01 = wmma32(a0, b1, acc01);
    acc10 = wmma32(a1, b0, acc10);
    acc11 = wmma32(a1, b1, acc11);
    if (kt + 1 < nk) {   // drain staged tile into the other LDS buffer
      const int nbuf = buf ^ 1;
      *(uint4*)&sA[nbuf][rA][cA] = ra;
      *(uint4*)&sB[nbuf][rB][cB] = rb0;
      *(uint4*)&sB[nbuf][rB][cB + 8] = rb1;
      __syncthreads();
    }
  }
  const int col = lane & 15, hi = lane >> 4;
  v8f accs[2][2] = {{acc00, acc01}, {acc10, acc11}};
#pragma unroll
  for (int sm = 0; sm < 2; ++sm)
#pragma unroll
    for (int sn = 0; sn < 2; ++sn) {
      int gn = n0 + wn + sn * 16 + col;
#pragma unroll
      for (int i = 0; i < 8; ++i) {
        int gm = m0 + wm + sm * 16 + i + 8 * hi;
        float v = accs[sm][sn][i];
        if (bias) v += bias[gn];
        if (xa) {
#pragma unroll
          for (int r = 0; r < kR; ++r)
            v += xa[(size_t)gm * kR + r] * lb[(size_t)r * N + gn];
        }
        Y[(size_t)gm * N + gn] = v;
      }
    }
}

// ---------------------------------------------------------------------------
// RoPE + f16 pack:  X f32 [S, nh, 64] -> f16 roped
// ---------------------------------------------------------------------------
__global__ void rope_pack(const float* __restrict__ X, const float* __restrict__ cb,
                          const float* __restrict__ sb, _Float16* __restrict__ out,
                          int nh, int total) {
  int i = blockIdx.x * 256 + threadIdx.x;
  if (i >= total) return;
  int d = i & 63;
  int s = i / (64 * nh);
  float v = X[i];
  float rot = (d < 32) ? -X[i + 32] : X[i - 32];
  out[i] = (_Float16)(v * cb[s * 64 + d] + rot * sb[s * 64 + d]);
}

__global__ void f32tof16(const float* __restrict__ X, _Float16* __restrict__ out, int total) {
  int i = blockIdx.x * 256 + threadIdx.x;
  if (i < total) out[i] = (_Float16)X[i];
}

// ---------------------------------------------------------------------------
// Flash attention: per-wave 16-row Q block, 32 keys per iter, all WMMA.
// Q f16 [S,32,64] (roped), K f16 [S,8,64], V f16 [S,8,64], O f16 [S,2048]
// ---------------------------------------------------------------------------
__global__ __launch_bounds__(128)
void flash_attn(const _Float16* __restrict__ Qh, const _Float16* __restrict__ Kh,
                const _Float16* __restrict__ Vh, _Float16* __restrict__ Oh,
                float* __restrict__ mbuf, float* __restrict__ lbuf) {
  __shared__ _Float16 sP[4][16][40];
  const int w = threadIdx.x >> 5, lane = threadIdx.x & 31;
  const int head = blockIdx.x >> 4;
  const int qblk = (blockIdx.x & 15) * 4 + w;
  const int qr0 = qblk * 16;
  const int kvh = head >> 2;                 // GQA: 4 q-heads per kv-head
  const int col = lane & 15, hi = lane >> 4;
  const _Float16* qbase = Qh + (size_t)qr0 * kC + head * kD;
  v16h aq0 = frag_a_mk(qbase, kC);
  v16h aq1 = frag_a_mk(qbase + 32, kC);
  float mrow[8], lrow[8];
  v8f ao[4];
  v8f zero = {};
#pragma unroll
  for (int i = 0; i < 8; ++i) { mrow[i] = -3.0e38f; lrow[i] = 0.0f; }
#pragma unroll
  for (int nd = 0; nd < 4; ++nd) ao[nd] = zero;
  const int kbmax = (qr0 + 15) >> 5;         // causal upper bound
  for (int kb = 0; kb <= kbmax; ++kb) {
    const int kbase = kb * 32;
    const _Float16* kp = Kh + (size_t)kbase * kKV + kvh * kD;
    v8f s0 = {}, s1 = {};
    {
      v16h b00 = frag_b_nk(kp, kKV);
      v16h b01 = frag_b_nk(kp + 32, kKV);
      s0 = wmma32(aq0, b00, s0);
      s0 = wmma32(aq1, b01, s0);
      v16h b10 = frag_b_nk(kp + (size_t)16 * kKV, kKV);
      v16h b11 = frag_b_nk(kp + (size_t)16 * kKV + 32, kKV);
      s1 = wmma32(aq0, b10, s1);
      s1 = wmma32(aq1, b11, s1);
    }
    float rmax[8], rsum[8];
#pragma unroll
    for (int i = 0; i < 8; ++i) {
      int qpos = qr0 + i + 8 * hi;
      float x0 = s0[i] * 0.125f + ((kbase + col) <= qpos ? 0.0f : -1e9f);
      float x1 = s1[i] * 0.125f + ((kbase + 16 + col) <= qpos ? 0.0f : -1e9f);
      s0[i] = x0; s1[i] = x1;
      rmax[i] = fmaxf(x0, x1);
    }
#pragma unroll
    for (int off = 1; off < 16; off <<= 1)
#pragma unroll
      for (int i = 0; i < 8; ++i)
        rmax[i] = fmaxf(rmax[i], __shfl_xor(rmax[i], off, 32));
#pragma unroll
    for (int i = 0; i < 8; ++i) {
      float nm = fmaxf(mrow[i], rmax[i]);
      float al = __expf(mrow[i] - nm);
      mrow[i] = nm;
      float p0 = __expf(s0[i] - nm), p1 = __expf(s1[i] - nm);
      rsum[i] = p0 + p1;
      lrow[i] *= al;
#pragma unroll
      for (int nd = 0; nd < 4; ++nd) ao[nd][i] *= al;
      sP[w][i + 8 * hi][col] = (_Float16)p0;
      sP[w][i + 8 * hi][16 + col] = (_Float16)p1;
    }
#pragma unroll
    for (int off = 1; off < 16; off <<= 1)
#pragma unroll
      for (int i = 0; i < 8; ++i)
        rsum[i] += __shfl_xor(rsum[i], off, 32);
#pragma unroll
    for (int i = 0; i < 8; ++i) lrow[i] += rsum[i];
    v16h ap = frag_a_mk(&sP[w][0][0], 40);   // intra-wave LDS RAW (dscnt wait)
    const _Float16* vb = Vh + (size_t)kbase * kKV + kvh * kD;
#pragma unroll
    for (int nd = 0; nd < 4; ++nd) {
      v16h bv = frag_b_gather(vb + nd * 16, kKV);
      ao[nd] = wmma32(ap, bv, ao[nd]);
    }
  }
#pragma unroll
  for (int i = 0; i < 8; ++i) {
    int qpos = qr0 + i + 8 * hi;
    float inv = 1.0f / lrow[i];
#pragma unroll
    for (int nd = 0; nd < 4; ++nd)
      Oh[(size_t)qpos * kC + head * kD + nd * 16 + col] = (_Float16)(ao[nd][i] * inv);
    if (col == 0) {
      mbuf[head * kS + qpos] = mrow[i];
      lbuf[head * kS + qpos] = lrow[i];
    }
  }
}

// ---------------------------------------------------------------------------
// Recompute QK^T (WMMA) and histogram softmax probs for the 0.95 quantile.
// ---------------------------------------------------------------------------
__global__ __launch_bounds__(128)
void attn_hist(const _Float16* __restrict__ Qh, const _Float16* __restrict__ Kh,
               const float* __restrict__ mbuf, const float* __restrict__ lbuf,
               unsigned int* __restrict__ hist) {
  __shared__ unsigned int hloc[2048];
  for (int i = threadIdx.x; i < 2048; i += 128) hloc[i] = 0;
  __syncthreads();
  const int w = threadIdx.x >> 5, lane = threadIdx.x & 31;
  const int head = blockIdx.x >> 4;
  const int qblk = (blockIdx.x & 15) * 4 + w;
  const int qr0 = qblk * 16;
  const int kvh = head >> 2;
  const int col = lane & 15, hi = lane >> 4;
  const _Float16* qbase = Qh + (size_t)qr0 * kC + head * kD;
  v16h aq0 = frag_a_mk(qbase, kC);
  v16h aq1 = frag_a_mk(qbase + 32, kC);
  float mr[8], li[8];
#pragma unroll
  for (int i = 0; i < 8; ++i) {
    int qpos = qr0 + i + 8 * hi;
    mr[i] = mbuf[head * kS + qpos];
    li[i] = 1.0f / lbuf[head * kS + qpos];
  }
  for (int kb = 0; kb < kS / 32; ++kb) {
    const int kbase = kb * 32;
    const _Float16* kp = Kh + (size_t)kbase * kKV + kvh * kD;
    v8f s0 = {}, s1 = {};
    v16h b00 = frag_b_nk(kp, kKV);
    v16h b01 = frag_b_nk(kp + 32, kKV);
    s0 = wmma32(aq0, b00, s0);
    s0 = wmma32(aq1, b01, s0);
    v16h b10 = frag_b_nk(kp + (size_t)16 * kKV, kKV);
    v16h b11 = frag_b_nk(kp + (size_t)16 * kKV + 32, kKV);
    s1 = wmma32(aq0, b10, s1);
    s1 = wmma32(aq1, b11, s1);
#pragma unroll
    for (int i = 0; i < 8; ++i) {
      int qpos = qr0 + i + 8 * hi;
      float p0 = (kbase + col) <= qpos ? __expf(s0[i] * 0.125f - mr[i]) * li[i] : 0.0f;
      float p1 = (kbase + 16 + col) <= qpos ? __expf(s1[i] * 0.125f - mr[i]) * li[i] : 0.0f;
      int b0 = (int)(p0 * 2048.0f); if (b0 > 2047) b0 = 2047;
      int b1 = (int)(p1 * 2048.0f); if (b1 > 2047) b1 = 2047;
      atomicAdd(&hloc[b0], 1u);
      atomicAdd(&hloc[b1], 1u);
    }
  }
  __syncthreads();
  for (int i = threadIdx.x; i < 2048; i += 128)
    if (hloc[i]) atomicAdd(&hist[i], hloc[i]);
}

__global__ void quantile95(const unsigned int* __restrict__ hist, float* __restrict__ out) {
  if (threadIdx.x != 0 || blockIdx.x != 0) return;
  unsigned long long total = (unsigned long long)kH * kS * kS;
  double target = 0.95 * (double)total;
  unsigned long long cum = 0;
  float thr = 1.0f;
  for (int b = 0; b < 2048; ++b) {
    cum += hist[b];
    if ((double)cum >= target) { thr = (b + 0.5f) / 2048.0f; break; }
  }
  out[0] = thr;
}

// ---------------------------------------------------------------------------
// Column-max reductions (deterministic, no float atomics).
// mode 0 : max|x| ; mode 1 : max|silu(x)|
// ---------------------------------------------------------------------------
__global__ void colmax_f32(const float* __restrict__ X, float* __restrict__ cm,
                           int rows, int C, int mode) {
  int c = blockIdx.x * 256 + threadIdx.x;
  if (c >= C) return;
  float m = 0.0f;
  for (int r = 0; r < rows; ++r) {
    float v = X[(size_t)r * C + c];
    if (mode == 1) v = v / (1.0f + __expf(-v));
    m = fmaxf(m, fabsf(v));
  }
  cm[c] = m;
}

__global__ void colmax_f16(const _Float16* __restrict__ X, float* __restrict__ cm,
                           int rows, int C) {
  int c = blockIdx.x * 256 + threadIdx.x;
  if (c >= C) return;
  float m = 0.0f;
  for (int r = 0; r < rows; ++r)
    m = fmaxf(m, fabsf((float)X[(size_t)r * C + c]));
  cm[c] = m;
}

// ---------------------------------------------------------------------------
// Elementwise: residual add, silu*up
// ---------------------------------------------------------------------------
__global__ void add_f32(const float* __restrict__ a, const float* __restrict__ b,
                        float* __restrict__ out, int total) {
  int i = blockIdx.x * 256 + threadIdx.x;
  if (i < total) out[i] = a[i] + b[i];
}

__global__ void silu_had(const float* __restrict__ gate, const float* __restrict__ up,
                         _Float16* __restrict__ had, int total) {
  int i = blockIdx.x * 256 + threadIdx.x;
  if (i >= total) return;
  float g = gate[i];
  float fn = g / (1.0f + __expf(-g));
  had[i] = (_Float16)(up[i] * fn);
}

// ---------------------------------------------------------------------------
// Top-20 channel selection (descending), idx as float + int
// ---------------------------------------------------------------------------
__global__ void topk20(const float* __restrict__ cm, int C,
                       float* __restrict__ outf, int* __restrict__ outi) {
  if (threadIdx.x != 0 || blockIdx.x != 0) return;
  for (int t = 0; t < kNO; ++t) {
    float best = -1.0f;
    int bi = 0;
    for (int c = 0; c < C; ++c) {
      bool used = false;
      for (int j = 0; j < t; ++j) used |= (outi[j] == c);
      float v = cm[c];
      if (!used && v > best) { best = v; bi = c; }
    }
    outi[t] = bi;
    outf[t] = (float)bi;
  }
}

// ---------------------------------------------------------------------------
// Final scale computation into d_out
// ---------------------------------------------------------------------------
__global__ void finalize_scales(float* __restrict__ out,
                                const float* cmx, const int* idxx,
                                const float* cmxn1, const float* cmq,
                                const float* cmk, const float* cmv,
                                const float* cmo, const float* cmxm,
                                const int* idxxm, const float* cmxn2,
                                const float* cmg, const float* cmu,
                                const float* cmfn, const float* cmhad) {
  int c = blockIdx.x * 256 + threadIdx.x;
  if (c >= kFF) return;
  if (c < kC) {
    bool o1 = false, o2 = false;
    for (int j = 0; j < kNO; ++j) { o1 |= (idxx[j] == c); o2 |= (idxxm[j] == c); }
    out[O_XSCALE + c]  = (o1 ? 0.0f : cmx[c]) * kQMaxInv + 1e-8f;
    out[O_XN1S + c]    = cmxn1[c] * kQMaxInv + 1e-8f;
    out[O_OS + c]      = cmo[c] * kQMaxInv + 1e-8f;
    out[O_XMSCALE + c] = (o2 ? 0.0f : cmxm[c]) * kQMaxInv + 1e-8f;
    out[O_XN2S + c]    = cmxn2[c] * kQMaxInv + 1e-8f;
  }
  if (c < kD) {
    out[O_QS + c] = cmq[c] * kQMaxInv + 1e-8f;
    out[O_KS + c] = cmk[c] * kQMaxInv + 1e-8f;
    out[O_VS + c] = cmv[c] * kQMaxInv + 1e-8f;
  }
  out[O_GS + c]   = cmg[c] * kQMaxInv + 1e-8f;
  out[O_US + c]   = cmu[c] * kQMaxInv + 1e-8f;
  out[O_FNS + c]  = cmfn[c] * kQMaxInv + 1e-8f;
  out[O_HADS + c] = cmhad[c] * kQMaxInv + 1e-8f;
}

// ---------------------------------------------------------------------------
// Host launcher
// ---------------------------------------------------------------------------
extern "C" void kernel_launch(void* const* d_in, const int* in_sizes, int n_in,
                              void* d_out, int out_size, void* d_ws, size_t ws_size,
                              hipStream_t stream) {
  (void)in_sizes; (void)n_in; (void)out_size; (void)ws_size;

  const float* x    = (const float*)d_in[0];
  const float* nw1  = (const float*)d_in[1];
  const float* nw2  = (const float*)d_in[2];
  const float* cosb = (const float*)d_in[3];
  const float* sinb = (const float*)d_in[4];
  // d_in[5] = mask: causal mask is generated in-kernel

  struct WSet { const int* wi; const float* sc; const float* b;
                const float* la; const float* lb; };
  auto getw = [&](int base) {
    WSet s{(const int*)d_in[base], (const float*)d_in[base + 1],
           (const float*)d_in[base + 2], (const float*)d_in[base + 3],
           (const float*)d_in[base + 4]};
    return s;
  };
  WSet Wq = getw(6), Wk = getw(11), Wv = getw(16), Wo = getw(21),
       Wg = getw(26), Wu = getw(31), Wd = getw(36);

  // workspace bump allocator
  size_t off = 0;
  auto take = [&](size_t bytes) -> void* {
    void* p = (char*)d_ws + off;
    off += (bytes + 255) & ~(size_t)255;
    return p;
  };
  _Float16* wq16 = (_Float16*)take((size_t)kC * kC * 2);
  _Float16* wk16 = (_Float16*)take((size_t)kKV * kC * 2);
  _Float16* wv16 = (_Float16*)take((size_t)kKV * kC * 2);
  _Float16* wo16 = (_Float16*)take((size_t)kC * kC * 2);
  _Float16* wg16 = (_Float16*)take((size_t)kFF * kC * 2);
  _Float16* wu16 = (_Float16*)take((size_t)kFF * kC * 2);
  _Float16* wd16 = (_Float16*)take((size_t)kC * kFF * 2);
  _Float16* xn1  = (_Float16*)take((size_t)kS * kC * 2);
  _Float16* xn2  = (_Float16*)take((size_t)kS * kC * 2);
  float* qf   = (float*)take((size_t)kS * kC * 4);
  float* kf   = (float*)take((size_t)kS * kKV * 4);
  float* vf   = (float*)take((size_t)kS * kKV * 4);
  _Float16* q16 = (_Float16*)take((size_t)kS * kC * 2);
  _Float16* k16 = (_Float16*)take((size_t)kS * kKV * 2);
  _Float16* v16buf = (_Float16*)take((size_t)kS * kKV * 2);
  _Float16* o16 = (_Float16*)take((size_t)kS * kC * 2);
  float* ofin  = (float*)take((size_t)kS * kC * 4);
  float* xmed  = (float*)take((size_t)kS * kC * 4);
  float* gatef = (float*)take((size_t)kS * kFF * 4);
  float* upf   = (float*)take((size_t)kS * kFF * 4);
  _Float16* had16 = (_Float16*)take((size_t)kS * kFF * 2);
  float* downf = (float*)take((size_t)kS * kC * 4);
  float* xaq = (float*)take((size_t)kS * kR * 4);
  float* xak = (float*)take((size_t)kS * kR * 4);
  float* xav = (float*)take((size_t)kS * kR * 4);
  float* xao = (float*)take((size_t)kS * kR * 4);
  float* xag = (float*)take((size_t)kS * kR * 4);
  float* xau = (float*)take((size_t)kS * kR * 4);
  float* xad = (float*)take((size_t)kS * kR * 4);
  float* mbuf = (float*)take((size_t)kH * kS * 4);
  float* lbuf = (float*)take((size_t)kH * kS * 4);
  unsigned int* hist = (unsigned int*)take(2048 * 4);
  float* cmx   = (float*)take(kC * 4);
  float* cmxn1 = (float*)take(kC * 4);
  float* cmq   = (float*)take(kD * 4);
  float* cmk   = (float*)take(kD * 4);
  float* cmv   = (float*)take(kD * 4);
  float* cmo   = (float*)take(kC * 4);
  float* cmxm  = (float*)take(kC * 4);
  float* cmxn2 = (float*)take(kC * 4);
  float* cmg   = (float*)take(kFF * 4);
  float* cmu   = (float*)take(kFF * 4);
  float* cmfn  = (float*)take(kFF * 4);
  float* cmhad = (float*)take(kFF * 4);
  int* idxx  = (int*)take(kNO * 4);
  int* idxxm = (int*)take(kNO * 4);

  float* outp = (float*)d_out;

  // 1) dequantize int32 weights -> f16
  dequant_w<<<2048, 256, 0, stream>>>(Wq.wi, Wq.sc, wq16, (size_t)kC * kC, kC);
  dequant_w<<<2048, 256, 0, stream>>>(Wk.wi, Wk.sc, wk16, (size_t)kKV * kC, kC);
  dequant_w<<<2048, 256, 0, stream>>>(Wv.wi, Wv.sc, wv16, (size_t)kKV * kC, kC);
  dequant_w<<<2048, 256, 0, stream>>>(Wo.wi, Wo.sc, wo16, (size_t)kC * kC, kC);
  dequant_w<<<2048, 256, 0, stream>>>(Wg.wi, Wg.sc, wg16, (size_t)kFF * kC, kC);
  dequant_w<<<2048, 256, 0, stream>>>(Wu.wi, Wu.sc, wu16, (size_t)kFF * kC, kC);
  dequant_w<<<2048, 256, 0, stream>>>(Wd.wi, Wd.sc, wd16, (size_t)kC * kFF, kFF);

  // 2) rmsnorm1 + input channel maxes
  rmsnorm_f16<<<kS, 256, 0, stream>>>(x, nw1, xn1, kC);
  colmax_f32<<<(kC + 255) / 256, 256, 0, stream>>>(x, cmx, kS, kC, 0);
  colmax_f16<<<(kC + 255) / 256, 256, 0, stream>>>(xn1, cmxn1, kS, kC);

  // 3) q/k/v projections (LoRA fused in GEMM epilogue)
  lora_xa<<<(kS * kR + 255) / 256, 256, 0, stream>>>(xn1, Wq.la, xaq, kS, kC);
  lora_xa<<<(kS * kR + 255) / 256, 256, 0, stream>>>(xn1, Wk.la, xak, kS, kC);
  lora_xa<<<(kS * kR + 255) / 256, 256, 0, stream>>>(xn1, Wv.la, xav, kS, kC);
  wmma_gemm_f16<<<dim3(kC / 128, kS / 64), 256, 0, stream>>>(
      xn1, wq16, Wq.b, xaq, Wq.lb, qf, kS, kC, kC);
  wmma_gemm_f16<<<dim3(kKV / 128, kS / 64), 256, 0, stream>>>(
      xn1, wk16, Wk.b, xak, Wk.lb, kf, kS, kKV, kC);
  wmma_gemm_f16<<<dim3(kKV / 128, kS / 64), 256, 0, stream>>>(
      xn1, wv16, Wv.b, xav, Wv.lb, vf, kS, kKV, kC);

  // 4) RoPE + f16 pack + head-dim maxes
  rope_pack<<<(kS * kC + 255) / 256, 256, 0, stream>>>(qf, cosb, sinb, q16, kH, kS * kC);
  rope_pack<<<(kS * kKV + 255) / 256, 256, 0, stream>>>(kf, cosb, sinb, k16, kHKV, kS * kKV);
  f32tof16<<<(kS * kKV + 255) / 256, 256, 0, stream>>>(vf, v16buf, kS * kKV);
  colmax_f16<<<1, 256, 0, stream>>>(q16, cmq, kS * kH, kD);
  colmax_f16<<<1, 256, 0, stream>>>(k16, cmk, kS * kHKV, kD);
  colmax_f16<<<1, 256, 0, stream>>>(v16buf, cmv, kS * kHKV, kD);

  // 5) flash attention + quantile of softmax probs
  flash_attn<<<512, 128, 0, stream>>>(q16, k16, v16buf, o16, mbuf, lbuf);
  hipMemsetAsync(hist, 0, 2048 * 4, stream);
  attn_hist<<<512, 128, 0, stream>>>(q16, k16, mbuf, lbuf, hist);
  quantile95<<<1, 1, 0, stream>>>(hist, outp + O_ATHR);
  colmax_f16<<<(kC + 255) / 256, 256, 0, stream>>>(o16, cmo, kS, kC);

  // 6) o projection + residual
  lora_xa<<<(kS * kR + 255) / 256, 256, 0, stream>>>(o16, Wo.la, xao, kS, kC);
  wmma_gemm_f16<<<dim3(kC / 128, kS / 64), 256, 0, stream>>>(
      o16, wo16, Wo.b, xao, Wo.lb, ofin, kS, kC, kC);
  add_f32<<<(kS * kC + 255) / 256, 256, 0, stream>>>(x, ofin, xmed, kS * kC);
  colmax_f32<<<(kC + 255) / 256, 256, 0, stream>>>(xmed, cmxm, kS, kC, 0);

  // 7) rmsnorm2 + FFN
  rmsnorm_f16<<<kS, 256, 0, stream>>>(xmed, nw2, xn2, kC);
  colmax_f16<<<(kC + 255) / 256, 256, 0, stream>>>(xn2, cmxn2, kS, kC);
  lora_xa<<<(kS * kR + 255) / 256, 256, 0, stream>>>(xn2, Wg.la, xag, kS, kC);
  lora_xa<<<(kS * kR + 255) / 256, 256, 0, stream>>>(xn2, Wu.la, xau, kS, kC);
  wmma_gemm_f16<<<dim3(kFF / 128, kS / 64), 256, 0, stream>>>(
      xn2, wg16, Wg.b, xag, Wg.lb, gatef, kS, kFF, kC);
  wmma_gemm_f16<<<dim3(kFF / 128, kS / 64), 256, 0, stream>>>(
      xn2, wu16, Wu.b, xau, Wu.lb, upf, kS, kFF, kC);
  silu_had<<<(kS * kFF + 255) / 256, 256, 0, stream>>>(gatef, upf, had16, kS * kFF);
  colmax_f32<<<(kFF + 255) / 256, 256, 0, stream>>>(gatef, cmg, kS, kFF, 0);
  colmax_f32<<<(kFF + 255) / 256, 256, 0, stream>>>(gatef, cmfn, kS, kFF, 1);
  colmax_f32<<<(kFF + 255) / 256, 256, 0, stream>>>(upf, cmu, kS, kFF, 0);
  colmax_f16<<<(kFF + 255) / 256, 256, 0, stream>>>(had16, cmhad, kS, kFF);

  // 8) down projection + final residual -> d_out
  lora_xa<<<(kS * kR + 255) / 256, 256, 0, stream>>>(had16, Wd.la, xad, kS, kFF);
  wmma_gemm_f16<<<dim3(kC / 128, kS / 64), 256, 0, stream>>>(
      had16, wd16, Wd.b, xad, Wd.lb, downf, kS, kC, kFF);
  add_f32<<<(kS * kC + 255) / 256, 256, 0, stream>>>(xmed, downf, outp, kS * kC);

  // 9) outlier indices + all quant scales
  topk20<<<1, 1, 0, stream>>>(cmx, kC, outp + O_XIDX, idxx);
  topk20<<<1, 1, 0, stream>>>(cmxm, kC, outp + O_XMIDX, idxxm);
  finalize_scales<<<(kFF + 255) / 256, 256, 0, stream>>>(
      outp, cmx, idxx, cmxn1, cmq, cmk, cmv, cmo, cmxm, idxxm, cmxn2,
      cmg, cmu, cmfn, cmhad);
}
